// SelfAttention_75153337745652
// MI455X (gfx1250) — matrix operand order repaired and checked by reading, not compiled, not run on hardware
//
#include <hip/hip_runtime.h>
#include <hip/hip_fp16.h>
#include <cstdint>

// ---------------------------------------------------------------------------
// Self-attention w/ ALiBi + tanh-cap, B=4 S=2048 D=2048 H=16 HD=128, fp32 I/O.
// f16 WMMA (16x16x32) everywhere, f32 accumulation.
// CDNA5 paths: async global->LDS tile staging (ASYNCcnt), double-buffered LDS.
// ---------------------------------------------------------------------------

typedef __attribute__((ext_vector_type(16))) _Float16 v16h;
typedef __attribute__((ext_vector_type(8)))  _Float16 v8h;
typedef __attribute__((ext_vector_type(4)))  _Float16 v4h;
typedef __attribute__((ext_vector_type(2)))  _Float16 v2h;
typedef __attribute__((ext_vector_type(2)))  __fp16   v2fp;
typedef __attribute__((ext_vector_type(8)))  float    v8f;

static constexpr int Bb  = 4;
static constexpr int Ss  = 2048;
static constexpr int Dd  = 2048;
static constexpr int Hh  = 16;
static constexpr int HDd = 128;

// ---- CDNA5 async global->LDS copy: 16B per lane, tracked by ASYNCcnt ----
__device__ __forceinline__ void async_copy16(unsigned ldsByteAddr, const _Float16* g) {
  unsigned long long ga = (unsigned long long)(uintptr_t)g;
  asm volatile("global_load_async_to_lds_b128 %0, %1, off"
               :: "v"(ldsByteAddr), "v"(ga) : "memory");
}

template <int N>
__device__ __forceinline__ void wait_async_le() {
#if __has_builtin(__builtin_amdgcn_s_wait_asynccnt)
  __builtin_amdgcn_s_wait_asynccnt((unsigned short)N);
#else
  asm volatile("s_wait_asynccnt %0" :: "i"(N) : "memory");
#endif
}

__device__ __forceinline__ unsigned lds_addr(const _Float16* p) {
  return (unsigned)(uintptr_t)p;  // low 32 bits of flat shared ptr = LDS offset
}

// ===========================================================================
// fp32 -> f16 elementwise converter (memory-bound pre-pass), float4 vectorized
// ===========================================================================
__global__ __launch_bounds__(256) void cvt_f32_to_f16(
    const float* __restrict__ in, _Float16* __restrict__ out, int n4)
{
  int i = blockIdx.x * 256 + threadIdx.x;
  if (i >= n4) return;
  float4 v = *(const float4*)(in + (size_t)i * 4);
  v2h lo = __builtin_bit_cast(v2h, __builtin_amdgcn_cvt_pkrtz(v.x, v.y));
  v2h hi = __builtin_bit_cast(v2h, __builtin_amdgcn_cvt_pkrtz(v.z, v.w));
  v4h p;
  p[0] = lo[0]; p[1] = lo[1]; p[2] = hi[0]; p[3] = hi[1];
  *(v4h*)(out + (size_t)i * 4) = p;
}

// ===========================================================================
// GEMM: out[m,n] = sum_k A[m,k] * W[n,k] + bias[n]   (A,W in f16, K-major rows)
// Tile 128x128x64, 256 threads = 8 waves (2x4), 16 WMMA/tile/wave.
// Double-buffered LDS filled by global_load_async_to_lds_b128.
// MODE 0: out -> f16 [B,H,S,HD].   MODE 1: out -> fp32 [M,N].
// ===========================================================================
template <int MODE>
__global__ __launch_bounds__(256) void gemm_xwT(
    const _Float16* __restrict__ A, const _Float16* __restrict__ W,
    const float* __restrict__ bias,
    _Float16* __restrict__ outH, float* __restrict__ outF)
{
  constexpr int BM = 128, BN = 128, BK = 64;
  constexpr int LDT = BK + 8;                 // 72 halves = 144B rows (16B-mult)
  constexpr int TILE = BM * LDT;              // halves per matrix tile
  __shared__ __align__(16) _Float16 smem[4 * TILE];  // [A0][B0][A1][B1] = 72 KB

  const int tid  = threadIdx.x;
  const int lane = tid & 31;
  const int wid  = tid >> 5;
  const int wm   = wid & 1;
  const int wn   = wid >> 1;
  const int hi   = lane >> 4;
  const int ln   = lane & 15;

  const int m0 = blockIdx.x * BM;
  const int n0 = blockIdx.y * BN;

  v8f acc[4][2];
#pragma unroll
  for (int i = 0; i < 4; ++i)
#pragma unroll
    for (int j = 0; j < 2; ++j)
#pragma unroll
      for (int e = 0; e < 8; ++e) acc[i][j][e] = 0.0f;

  // issue async copies for one K-tile into buffer p (8 instructions/thread)
  auto issue_tile = [&](int p, int k0) {
    _Float16* Ap = smem + p * 2 * TILE;
    _Float16* Bp = Ap + TILE;
#pragma unroll
    for (int i = 0; i < 4; ++i) {
      int linear = tid + i * 256;          // 1024 chunks of 8 halves (16B)
      int r = linear >> 3;
      int c = (linear & 7) << 3;
      async_copy16(lds_addr(Ap + r * LDT + c), A + (size_t)(m0 + r) * Dd + k0 + c);
      async_copy16(lds_addr(Bp + r * LDT + c), W + (size_t)(n0 + r) * Dd + k0 + c);
    }
  };

  issue_tile(0, 0);

  for (int k0 = 0; k0 < Dd; k0 += BK) {
    const int p = (k0 >> 6) & 1;
    if (k0 + BK < Dd) {
      issue_tile(p ^ 1, k0 + BK);
      wait_async_le<8>();   // oldest 8 (current tile) done; next 8 in flight
    } else {
      wait_async_le<0>();
    }
    __syncthreads();

    const _Float16* At = smem + p * 2 * TILE;
    const _Float16* Bt = At + TILE;

#pragma unroll
    for (int kk = 0; kk < BK; kk += 32) {
      v16h bfr[2];
#pragma unroll
      for (int j = 0; j < 2; ++j) {
        const _Float16* br = Bt + (wn * 32 + j * 16 + ln) * LDT + kk;
        v8h b0 = *(const v8h*)(br + hi * 8);
        v8h b1 = *(const v8h*)(br + 16 + hi * 8);
#pragma unroll
        for (int e = 0; e < 8; ++e) { bfr[j][e] = b0[e]; bfr[j][8 + e] = b1[e]; }
      }
#pragma unroll
      for (int i = 0; i < 4; ++i) {
        const _Float16* ar = At + (wm * 64 + i * 16 + ln) * LDT + kk;
        v8h a0 = *(const v8h*)(ar + hi * 8);
        v8h a1 = *(const v8h*)(ar + 16 + hi * 8);
        v16h af;
#pragma unroll
        for (int e = 0; e < 8; ++e) { af[e] = a0[e]; af[8 + e] = a1[e]; }
#pragma unroll
        for (int j = 0; j < 2; ++j)
          acc[i][j] = __builtin_amdgcn_wmma_f32_16x16x32_f16(
              false, af, false, bfr[j], (short)0, acc[i][j], false, false);
      }
    }
    __syncthreads();
  }

  // ---- epilogue ----
#pragma unroll
  for (int i = 0; i < 4; ++i) {
    int mBase = m0 + wm * 64 + i * 16 + hi * 8;
#pragma unroll
    for (int j = 0; j < 2; ++j) {
      int n = n0 + wn * 32 + j * 16 + ln;
      float bn = bias[n];
#pragma unroll
      for (int e = 0; e < 8; ++e) {
        float val = acc[i][j][e] + bn;
        int gm = mBase + e;
        if (MODE == 0) {
          int b = gm >> 11, s = gm & 2047;
          int h = n >> 7,  hd = n & 127;
          outH[((size_t)(b * Hh + h) * Ss + s) * HDd + hd] = (_Float16)val;
        } else {
          outF[(size_t)gm * Dd + n] = val;
        }
      }
    }
  }
}

// ===========================================================================
// Flash attention: grid (S/64, H, B), 128 threads = 4 waves.
// Wave w owns query rows [q0, q0+16); key blocks of 32; online softmax.
// ===========================================================================
__global__ __launch_bounds__(128) void attn_kernel(
    const _Float16* __restrict__ Qh, const _Float16* __restrict__ Kh,
    const _Float16* __restrict__ Vh, const float* __restrict__ slopes,
    _Float16* __restrict__ Ctx)
{
  constexpr int KT  = 32;
  constexpr int LDV = KT + 8;  // 40 halves -> 80B rows
  __shared__ __align__(16) _Float16 Vt[HDd][LDV];    // V^T block
  __shared__ __align__(16) _Float16 Pt[4][16][LDV];  // per-wave P staging

  const int tid  = threadIdx.x;
  const int lane = tid & 31;
  const int w    = tid >> 5;
  const int hi   = lane >> 4;
  const int ln   = lane & 15;
  const int b    = blockIdx.z;
  const int h    = blockIdx.y;
  const int q0blk = blockIdx.x * 64;
  const int q0    = q0blk + w * 16;
  const float slope = slopes[h];
  const float inv_sqrt_hd = 0.08838834764831845f;  // 1/sqrt(128)

  const size_t headOff = (size_t)(b * Hh + h) * Ss * HDd;
  const _Float16* Qp = Qh + headOff;
  const _Float16* Kp = Kh + headOff;
  const _Float16* Vp = Vh + headOff;

  // ---- Q fragments: 4 chunks of K=32, row = q0+ln ----
  v16h qf[4];
  {
    const _Float16* qrow = Qp + (size_t)(q0 + ln) * HDd;
#pragma unroll
    for (int c = 0; c < 4; ++c) {
      v8h a0 = *(const v8h*)(qrow + c * 32 + hi * 8);
      v8h a1 = *(const v8h*)(qrow + c * 32 + 16 + hi * 8);
#pragma unroll
      for (int e = 0; e < 8; ++e) { qf[c][e] = a0[e]; qf[c][8 + e] = a1[e]; }
    }
  }

  v8f o[8];
#pragma unroll
  for (int fc = 0; fc < 8; ++fc)
#pragma unroll
    for (int e = 0; e < 8; ++e) o[fc][e] = 0.0f;
  float Mr[8], Lr[8];
#pragma unroll
  for (int i = 0; i < 8; ++i) { Mr[i] = -1e30f; Lr[i] = 0.0f; }

  const int kbMax = (q0blk + 63) / KT;
  for (int kb = 0; kb <= kbMax; ++kb) {
    const int kbase = kb * KT;

    // ---- stage V block transposed: Vt[feat][key] ----
#pragma unroll
    for (int i = 0; i < 4; ++i) {
      int linear = tid + i * 128;            // 512 chunks of 8 halves
      int key = linear >> 4;
      int c8  = (linear & 15) << 3;
      v8h v = *(const v8h*)(Vp + (size_t)(kbase + key) * HDd + c8);
#pragma unroll
      for (int e = 0; e < 8; ++e) Vt[c8 + e][key] = v[e];
    }
    __syncthreads();

    // ---- scores: S = Q K^T over two 16-key subtiles ----
    v8f sc[2];
#pragma unroll
    for (int j = 0; j < 2; ++j) {
#pragma unroll
      for (int e = 0; e < 8; ++e) sc[j][e] = 0.0f;
      const _Float16* krow = Kp + (size_t)(kbase + j * 16 + ln) * HDd;
#pragma unroll
      for (int c = 0; c < 4; ++c) {
        v8h b0 = *(const v8h*)(krow + c * 32 + hi * 8);
        v8h b1 = *(const v8h*)(krow + c * 32 + 16 + hi * 8);
        v16h kf;
#pragma unroll
        for (int e = 0; e < 8; ++e) { kf[e] = b0[e]; kf[8 + e] = b1[e]; }
        sc[j] = __builtin_amdgcn_wmma_f32_16x16x32_f16(
            false, qf[c], false, kf, (short)0, sc[j], false, false);
      }
    }

    // ---- alibi + scale + tanh cap + causal mask ----
#pragma unroll
    for (int i = 0; i < 8; ++i) {
      int qrow = q0 + hi * 8 + i;
#pragma unroll
      for (int j = 0; j < 2; ++j) {
        int key = kbase + j * 16 + ln;
        float s = sc[j][i];
        s = (s + slope * (float)(key - qrow)) * inv_sqrt_hd;
        s = 30.0f * tanhf(s * (1.0f / 30.0f));
        sc[j][i] = (key <= qrow) ? s : -1e30f;
      }
    }

    // ---- online softmax (row = hi*8+i, cols across 16-lane halves) ----
#pragma unroll
    for (int i = 0; i < 8; ++i) {
      float rmax = fmaxf(sc[0][i], sc[1][i]);
#pragma unroll
      for (int m = 8; m >= 1; m >>= 1)
        rmax = fmaxf(rmax, __shfl_xor(rmax, m, 32));
      float Mnew  = fmaxf(Mr[i], rmax);
      float scale = __expf(Mr[i] - Mnew);
      Mr[i] = Mnew;
      float p0 = (sc[0][i] > -5e29f) ? __expf(sc[0][i] - Mnew) : 0.0f;
      float p1 = (sc[1][i] > -5e29f) ? __expf(sc[1][i] - Mnew) : 0.0f;
      float psum = p0 + p1;
#pragma unroll
      for (int m = 8; m >= 1; m >>= 1)
        psum += __shfl_xor(psum, m, 32);
      Lr[i] = Lr[i] * scale + psum;
#pragma unroll
      for (int fc = 0; fc < 8; ++fc) o[fc][i] *= scale;
      Pt[w][hi * 8 + i][ln]      = (_Float16)p0;
      Pt[w][hi * 8 + i][16 + ln] = (_Float16)p1;
    }
    __syncthreads();

    // ---- O += P V : P is A-matrix (16x32), V^T rows are B-fragments ----
    v16h pf;
    {
      const _Float16* prow = &Pt[w][ln][0];
      v8h a0 = *(const v8h*)(prow + hi * 8);
      v8h a1 = *(const v8h*)(prow + 16 + hi * 8);
#pragma unroll
      for (int e = 0; e < 8; ++e) { pf[e] = a0[e]; pf[8 + e] = a1[e]; }
    }
#pragma unroll
    for (int fc = 0; fc < 8; ++fc) {
      const _Float16* vrow = &Vt[fc * 16 + ln][0];
      v8h b0 = *(const v8h*)(vrow + hi * 8);
      v8h b1 = *(const v8h*)(vrow + 16 + hi * 8);
      v16h vf;
#pragma unroll
      for (int e = 0; e < 8; ++e) { vf[e] = b0[e]; vf[8 + e] = b1[e]; }
      o[fc] = __builtin_amdgcn_wmma_f32_16x16x32_f16(
          false, pf, false, vf, (short)0, o[fc], false, false);
    }
    __syncthreads();
  }

  // ---- normalize + write ctx in [B,S,H*HD] f16 ----
#pragma unroll
  for (int fc = 0; fc < 8; ++fc) {
#pragma unroll
    for (int i = 0; i < 8; ++i) {
      int qrow = q0 + hi * 8 + i;
      float val = o[fc][i] / Lr[i];
      Ctx[((size_t)b * Ss + qrow) * Dd + h * HDd + fc * 16 + ln] = (_Float16)val;
    }
  }
}

// ===========================================================================
extern "C" void kernel_launch(void* const* d_in, const int* in_sizes, int n_in,
                              void* d_out, int out_size, void* d_ws, size_t ws_size,
                              hipStream_t stream) {
  const float* X      = (const float*)d_in[0];
  const float* Wq     = (const float*)d_in[1];
  const float* bq     = (const float*)d_in[2];
  const float* Wk     = (const float*)d_in[3];
  const float* bk     = (const float*)d_in[4];
  const float* Wv     = (const float*)d_in[5];
  const float* bv     = (const float*)d_in[6];
  const float* Wo     = (const float*)d_in[7];
  const float* bo     = (const float*)d_in[8];
  const float* slopes = (const float*)d_in[9];

  const size_t nX = (size_t)Bb * Ss * Dd;   // 16.7M elements
  const size_t nW = (size_t)Dd * Dd;        // 4.2M elements
  _Float16* Xh  = (_Float16*)d_ws;
  _Float16* Wqh = Xh  + nX;
  _Float16* Wkh = Wqh + nW;
  _Float16* Wvh = Wkh + nW;
  _Float16* Woh = Wvh + nW;
  _Float16* Qh  = Woh + nW;
  _Float16* Kh  = Qh + nX;
  _Float16* Vh  = Kh + nX;
  _Float16* Ctx = Vh + nX;

  // ---- pre-pass: fp32 -> f16 (memory-bound, done once) ----
  int n4x = (int)(nX / 4), n4w = (int)(nW / 4);
  cvt_f32_to_f16<<<(n4x + 255) / 256, 256, 0, stream>>>(X,  Xh,  n4x);
  cvt_f32_to_f16<<<(n4w + 255) / 256, 256, 0, stream>>>(Wq, Wqh, n4w);
  cvt_f32_to_f16<<<(n4w + 255) / 256, 256, 0, stream>>>(Wk, Wkh, n4w);
  cvt_f32_to_f16<<<(n4w + 255) / 256, 256, 0, stream>>>(Wv, Wvh, n4w);
  cvt_f32_to_f16<<<(n4w + 255) / 256, 256, 0, stream>>>(Wo, Woh, n4w);

  dim3 ggrid(Bb * Ss / 128, Dd / 128);  // (64,16)
  dim3 gblk(256);
  gemm_xwT<0><<<ggrid, gblk, 0, stream>>>(Xh, Wqh, bq, Qh, nullptr);
  gemm_xwT<0><<<ggrid, gblk, 0, stream>>>(Xh, Wkh, bk, Kh, nullptr);
  gemm_xwT<0><<<ggrid, gblk, 0, stream>>>(Xh, Wvh, bv, Vh, nullptr);

  attn_kernel<<<dim3(Ss / 64, Hh, Bb), 128, 0, stream>>>(Qh, Kh, Vh, slopes, Ctx);

  gemm_xwT<1><<<ggrid, gblk, 0, stream>>>(Ctx, Woh, bo, nullptr, (float*)d_out);
}
